// SelfAttention_50311246905875
// MI455X (gfx1250) — matrix operand order repaired and checked by reading, not compiled, not run on hardware
//
#include <hip/hip_runtime.h>

// ---------------------------------------------------------------------------
// Flash-attention for B=4, H=16, S=2048, dh=64, fp32 I/O, multiplicative mask.
// bf16 WMMA (v_wmma_f32_16x16x32_bf16) with fp32 accumulation, wave32.
// 1 block = 128 threads = 4 waves; block owns (b, h, 64-query tile);
// each wave owns 16 query rows. Key blocks of 64 staged in LDS as bf16.
// Softmax runs in the log2 domain: scale*log2(e) folded into one constant,
// so each score element costs mul + sub + fma + v_exp_f32.
// ---------------------------------------------------------------------------

typedef __attribute__((ext_vector_type(16))) __bf16 v16bf;
typedef __attribute__((ext_vector_type(8)))  __bf16 v8bf;
typedef __attribute__((ext_vector_type(8)))  float  v8f;

#define ATT_S   2048
#define ATT_H   16
#define ATT_DH  64
#define ATT_QT  64            // queries per block (16 per wave)
#define ATT_KT  64            // keys per iteration
#define VT_STRIDE 72          // padded V^T row stride (elems): 144B, 16B-aligned
#define ATT_C1  0.1803368801f // (1/sqrt(64)) * log2(e)
#define ATT_C2  1.442695041e-8f // eps * log2(e)

__device__ __forceinline__ v16bf join16(v8bf lo, v8bf hi) {
    return __builtin_shufflevector(lo, hi, 0,1,2,3,4,5,6,7,8,9,10,11,12,13,14,15);
}

__global__ void __launch_bounds__(128)
attn_fwd_wmma(const float* __restrict__ x,
              const float* __restrict__ mask,
              float* __restrict__ out) {
    // LDS: K row-major [key][d] bf16 (8KB), V^T padded [d][key] bf16 (9KB),
    //      P per-wave [16][64] bf16 (8KB)  -> 25KB total
    __shared__ __align__(32) __bf16 k_lds[ATT_KT * ATT_DH];
    __shared__ __align__(32) __bf16 vT_lds[ATT_DH * VT_STRIDE];
    __shared__ __align__(32) __bf16 p_lds[4][16 * ATT_KT];

    const int tid  = threadIdx.x;
    const int lane = tid & 31;
    const int wave = tid >> 5;           // 0..3
    const int col  = lane & 15;          // n within a 16-wide tile
    const int hi   = lane >> 4;          // 0/1 half-wave

    const int bid   = blockIdx.x;
    const int qtile = bid & 31;                  // S/QT = 32
    const int h     = (bid >> 5) & (ATT_H - 1);
    const int b     = bid >> 9;
    const int q0    = qtile * ATT_QT;

    const size_t row_stride = 3 * ATT_H * ATT_DH;          // 3072 floats
    const float* xb = x + (size_t)b * ATT_S * row_stride;
    const float* mb = mask + (size_t)b * ATT_S * ATT_S;

    // ---- Q A-fragments (resident all kernel). 16-bit A layout:
    // lane row = lane%16; halfword j<8 -> k = hi*8+j ; j>=8 -> k = 16+hi*8+(j-8).
    const int qrow = q0 + wave * 16 + col;
    const float* qp = xb + (size_t)qrow * row_stride + h * ATT_DH;
    v16bf a0, a1;
#pragma unroll
    for (int j = 0; j < 8; ++j) {
        a0[j]     = (__bf16)qp[hi * 8 + j];
        a0[j + 8] = (__bf16)qp[16 + hi * 8 + j];
        a1[j]     = (__bf16)qp[32 + hi * 8 + j];
        a1[j + 8] = (__bf16)qp[48 + hi * 8 + j];
    }

    float m_i[8], l_i[8];
#pragma unroll
    for (int r = 0; r < 8; ++r) { m_i[r] = -1e30f; l_i[r] = 0.0f; }
    v8f oacc[4];
#pragma unroll
    for (int t = 0; t < 4; ++t) oacc[t] = (v8f){};

    const v8f zero_c = (v8f){};
    __bf16* pw = &p_lds[wave][0];

    for (int k0 = 0; k0 < ATT_S; k0 += ATT_KT) {
        __syncthreads();   // previous iteration done reading K/V tiles

        // ---- cooperative staging: K tile row-major bf16, V tile transposed ----
        {
            const float* kp = xb + (size_t)k0 * row_stride + ATT_H * ATT_DH + h * ATT_DH;
#pragma unroll
            for (int e = 0; e < 16; ++e) {
                int idx = e * 256 + tid * 2;          // 4096 elems, pairs
                int key = idx >> 6, d = idx & 63;
                float2 f = *(const float2*)(kp + (size_t)key * row_stride + d);
                k_lds[idx]     = (__bf16)f.x;
                k_lds[idx + 1] = (__bf16)f.y;
            }
            const float* vp = kp + ATT_H * ATT_DH;
#pragma unroll
            for (int e = 0; e < 16; ++e) {
                int idx = e * 256 + tid * 2;
                int key = idx >> 6, d = idx & 63;     // d even
                float2 f = *(const float2*)(vp + (size_t)key * row_stride + d);
                vT_lds[d * VT_STRIDE + key]       = (__bf16)f.x;
                vT_lds[(d + 1) * VT_STRIDE + key] = (__bf16)f.y;
            }
        }

        // ---- prefetch next K/V tile (global_prefetch_b8, no reg cost) ----
        if (k0 + ATT_KT < ATT_S) {
            const float* np = xb + (size_t)(k0 + ATT_KT + (tid & 63)) * row_stride
                            + (1 + (tid >> 6)) * (ATT_H * ATT_DH) + h * ATT_DH;
            __builtin_prefetch(np, 0, 1);
            __builtin_prefetch(np + 32, 0, 1);
        }

        __syncthreads();   // tiles visible to all 4 waves

        // ---- S = Q * K^T : 4 score tiles (16 keys each), two dh halves ----
        v8f c[4];
#pragma unroll
        for (int t2 = 0; t2 < 4; ++t2) {
            const int kb0 = (col + 16 * t2) * ATT_DH + hi * 8;
            v16bf b0 = join16(*(const v8bf*)(k_lds + kb0),      *(const v8bf*)(k_lds + kb0 + 16));
            v16bf b1 = join16(*(const v8bf*)(k_lds + kb0 + 32), *(const v8bf*)(k_lds + kb0 + 48));
            c[t2] = __builtin_amdgcn_wmma_f32_16x16x32_bf16(false, a0, false, b0, (short)0, zero_c, false, false);
            c[t2] = __builtin_amdgcn_wmma_f32_16x16x32_bf16(false, a1, false, b1, (short)0, c[t2], false, false);
        }

        // ---- scale + multiplicative mask, in log2 domain ----
        // w2 = log2e*(mask*s*scale + (1-mask)*eps) = mask*(s*C1 - C2) + C2
#pragma unroll
        for (int r = 0; r < 8; ++r) {
            size_t moff = (size_t)(q0 + wave * 16 + r + 8 * hi) * ATT_S + k0 + col;
#pragma unroll
            for (int t2 = 0; t2 < 4; ++t2) {
                float mk = mb[moff + 16 * t2];
                c[t2][r] = fmaf(mk, c[t2][r] * ATT_C1 - ATT_C2, ATT_C2);
            }
        }

        // ---- online softmax over this 64-key block (exp2 domain) ----
        float corr[8];
#pragma unroll
        for (int r = 0; r < 8; ++r) {
            float mx = fmaxf(fmaxf(c[0][r], c[1][r]), fmaxf(c[2][r], c[3][r]));
#pragma unroll
            for (int s = 1; s < 16; s <<= 1) mx = fmaxf(mx, __shfl_xor(mx, s, 32));
            float mnew = fmaxf(m_i[r], mx);
            corr[r] = exp2f(m_i[r] - mnew);
            m_i[r]  = mnew;
            float rs = 0.0f;
#pragma unroll
            for (int t2 = 0; t2 < 4; ++t2) {
                c[t2][r] = exp2f(c[t2][r] - mnew);
                rs += c[t2][r];
            }
#pragma unroll
            for (int s = 1; s < 16; s <<= 1) rs += __shfl_xor(rs, s, 32);
            l_i[r] = l_i[r] * corr[r] + rs;
        }
#pragma unroll
        for (int t = 0; t < 4; ++t)
#pragma unroll
            for (int r = 0; r < 8; ++r) oacc[t][r] *= corr[r];

        // ---- C-layout -> A-layout for P via per-wave LDS tile ----
#pragma unroll
        for (int r = 0; r < 8; ++r) {
            int rr = (r + 8 * hi) * ATT_KT + col;
#pragma unroll
            for (int t2 = 0; t2 < 4; ++t2)
                pw[rr + 16 * t2] = (__bf16)c[t2][r];
        }
        asm volatile("s_wait_dscnt 0" ::: "memory");  // DS RAW: P stores -> P reads

        const int poff = col * ATT_KT + hi * 8;
        v16bf pa0 = join16(*(const v8bf*)(pw + poff),      *(const v8bf*)(pw + poff + 16));
        v16bf pa1 = join16(*(const v8bf*)(pw + poff + 32), *(const v8bf*)(pw + poff + 48));

        // ---- O += P * V : 4 output d-tiles x 2 key halves ----
#pragma unroll
        for (int t = 0; t < 4; ++t) {
            const int vbase = (16 * t + col) * VT_STRIDE + hi * 8;
            v16bf vb0 = join16(*(const v8bf*)(vT_lds + vbase),      *(const v8bf*)(vT_lds + vbase + 16));
            v16bf vb1 = join16(*(const v8bf*)(vT_lds + vbase + 32), *(const v8bf*)(vT_lds + vbase + 48));
            oacc[t] = __builtin_amdgcn_wmma_f32_16x16x32_bf16(false, pa0, false, vb0, (short)0, oacc[t], false, false);
            oacc[t] = __builtin_amdgcn_wmma_f32_16x16x32_bf16(false, pa1, false, vb1, (short)0, oacc[t], false, false);
        }
    }

    // ---- normalize and store (out[b][q][h*64+d], fp32) ----
#pragma unroll
    for (int r = 0; r < 8; ++r) {
        int qg = q0 + wave * 16 + r + 8 * hi;
        float inv = 1.0f / l_i[r];
        float* op = out + ((size_t)b * ATT_S + qg) * (ATT_H * ATT_DH) + h * ATT_DH + col;
#pragma unroll
        for (int t = 0; t < 4; ++t) op[16 * t] = oacc[t][r] * inv;
    }
}

extern "C" void kernel_launch(void* const* d_in, const int* in_sizes, int n_in,
                              void* d_out, int out_size, void* d_ws, size_t ws_size,
                              hipStream_t stream) {
    (void)in_sizes; (void)n_in; (void)out_size; (void)d_ws; (void)ws_size;
    const float* x    = (const float*)d_in[0];
    const float* mask = (const float*)d_in[1];
    float* out        = (float*)d_out;
    // grid: B * H * (S / 64) = 4 * 16 * 32 = 2048 blocks of 128 threads (4 waves)
    attn_fwd_wmma<<<dim3(2048), dim3(128), 0, stream>>>(x, mask, out);
}
